// MambaLM_43250320670964
// MI455X (gfx1250) — matrix operand order repaired and checked by reading, not compiled, not run on hardware
//
#include <hip/hip_runtime.h>

typedef __attribute__((ext_vector_type(16))) _Float16 v16h;
typedef __attribute__((ext_vector_type(8)))  _Float16 v8h;
typedef __attribute__((ext_vector_type(8)))  float    v8f;
typedef __attribute__((ext_vector_type(4)))  int      v4i;

#define V_  50257
#define D_  768
#define NL_ 4
#define DI_ 1536
#define DS_ 16
#define DC_ 4
#define DR_ 48
#define B_  2
#define L_  1024
#define M_  (B_*L_)        // 2048 token rows
#define XDBW_ (DR_+2*DS_)  // 80
#define NC_ 16             // scan chunks
#define LC_ (L_/NC_)       // 64 steps per chunk

// ---- CDNA5 async global->LDS copy (ASYNCcnt path), with safe fallback ----
#if defined(__AMDGCN__) && __has_builtin(__builtin_amdgcn_global_load_async_to_lds_b128)
#define HAVE_ASYNC 1
#else
#define HAVE_ASYNC 0
#endif

#if HAVE_ASYNC
// builtin signature (from compiler diagnostic): (v4i as1*, v4i as3*, Ii, Ii)
typedef __attribute__((address_space(1))) v4i* as1_v4i;
typedef __attribute__((address_space(3))) v4i* as3_v4i;
// flat->LDS: LDS aperture maps addr[31:0] to the LDS offset (ISA 00_overview
// aperture table), so truncation is the correct conversion.
__device__ __forceinline__ void async_cp16(_Float16* lp, const _Float16* gp) {
    __builtin_amdgcn_global_load_async_to_lds_b128(
        (as1_v4i)(unsigned long long)(const void*)gp,
        (as3_v4i)(unsigned)(unsigned long long)(void*)lp, 0, 0);
}
__device__ __forceinline__ void async_wait0() {
#if __has_builtin(__builtin_amdgcn_s_wait_asynccnt)
    __builtin_amdgcn_s_wait_asynccnt(0);
#else
    asm volatile("s_wait_asynccnt 0x0" ::: "memory");
#endif
}
#endif

// ---------------------------------------------------------------- embedding
__global__ void k_gather(const int* __restrict__ ids, const float* __restrict__ embed,
                         float* __restrict__ x, int n) {
    int i = blockIdx.x * blockDim.x + threadIdx.x;
    if (i >= n) return;
    int m = i / D_, c = i % D_;
    x[i] = embed[(long)ids[m] * D_ + c];
}

// ------------------------------------------------- strided f32 -> f16 cast
__global__ void k_cast16(const float* __restrict__ src, _Float16* __restrict__ dst,
                         long rows, int cols, int sld, int dld) {
    long i = (long)blockIdx.x * blockDim.x + threadIdx.x;
    long n = rows * (long)cols;
    if (i >= n) return;
    long r = i / cols; int c = (int)(i % cols);
    dst[r * dld + c] = (_Float16)src[r * sld + c];
}

// ------------------------------------------------------- LayerNorm -> f16
__global__ __launch_bounds__(256) void k_ln16(const float* __restrict__ x,
                                              const float* __restrict__ g,
                                              const float* __restrict__ b,
                                              _Float16* __restrict__ out) {
    int row = blockIdx.x;
    int tid = threadIdx.x;
    __shared__ float red[256];
    __shared__ float s_mean, s_rstd;
    const float* xr = x + (long)row * D_;
    float a0 = xr[tid], a1 = xr[tid + 256], a2 = xr[tid + 512];
    red[tid] = a0 + a1 + a2;
    __syncthreads();
    for (int s = 128; s > 0; s >>= 1) { if (tid < s) red[tid] += red[tid + s]; __syncthreads(); }
    if (tid == 0) s_mean = red[0] * (1.0f / D_);
    __syncthreads();
    float m = s_mean;
    float d0 = a0 - m, d1 = a1 - m, d2 = a2 - m;
    red[tid] = d0*d0 + d1*d1 + d2*d2;
    __syncthreads();
    for (int s = 128; s > 0; s >>= 1) { if (tid < s) red[tid] += red[tid + s]; __syncthreads(); }
    if (tid == 0) s_rstd = rsqrtf(red[0] * (1.0f / D_) + 1e-5f);
    __syncthreads();
    float r = s_rstd;
    long base = (long)row * D_;
    out[base + tid      ] = (_Float16)(d0 * r * g[tid      ] + b[tid      ]);
    out[base + tid + 256] = (_Float16)(d1 * r * g[tid + 256] + b[tid + 256]);
    out[base + tid + 512] = (_Float16)(d2 * r * g[tid + 512] + b[tid + 512]);
}

// -------------------------------------------------------- WMMA f16 GEMM
// C(MxN,f32) = A(MxK,f16,row) @ B; b_nt==0: B is KxN row-major,
// b_nt==1: B is NxK row-major (embed table used directly as B^T).
// 256 threads = 8 wave32. Block tile 128x64, K-step 32.
// Each wave: 32x32 register tile = 4 accumulators, 4 WMMA per K-step
// (A and B fragments each reused twice -> half the LDS traffic per FLOP).
// Fast path is block-uniform; A and NT-B staged via async global->LDS.
#define BM 128
#define BN 64
#define BK 32
__global__ __launch_bounds__(256) void k_gemm_wmma(
        const _Float16* __restrict__ A, const _Float16* __restrict__ Bm,
        float* __restrict__ C,
        int M, int N, int K, int lda, int ldb, int ldc, int b_nt) {
    __shared__ __align__(16) _Float16 As[BM * 40];   // [m][k], padded stride 40
    __shared__ __align__(16) _Float16 Bs[BN * 40];   // [n][k], padded stride 40
    const int tid  = threadIdx.x;
    const int lane = tid & 31;
    const int wave = tid >> 5;
    const int wm   = wave & 3;   // 4 m-tiles of 32
    const int wn   = wave >> 2;  // 2 n-tiles of 32
    const int m0 = blockIdx.y * BM, n0 = blockIdx.x * BN;

    v8f c00 = {}; v8f c01 = {}; v8f c10 = {}; v8f c11 = {};

    for (int k0 = 0; k0 < K; k0 += BK) {
        // ---- stage A tile: 128 rows x 32 k; 16 f16 (32B) per thread
        {
            const int arow = tid >> 1;
            const int acg  = (tid & 1) << 4;
            const _Float16* agp = A + (long)(m0 + arow) * lda + k0 + acg;
            _Float16* alp = &As[arow * 40 + acg];
            if ((m0 + BM <= M) && (k0 + BK <= K)) {        // block-uniform fast path
#if HAVE_ASYNC
                async_cp16(alp, agp);
                async_cp16(alp + 8, agp + 8);
#else
                *(v8h*)alp       = *(const v8h*)agp;
                *(v8h*)(alp + 8) = *(const v8h*)(agp + 8);
#endif
                __builtin_prefetch(agp + BK, 0, 1);        // next K-tile into L2
            } else {
                for (int j = 0; j < 16; ++j)
                    alp[j] = ((m0 + arow) < M && (k0 + acg + j) < K) ? agp[j] : (_Float16)0.f;
            }
        }
        // ---- stage B tile into transposed LDS layout Bs[n][k]
        if (!b_nt) {  // B KxN row-major: vector-load a row chunk, scatter transposed
            const int kk = tid >> 3;            // 0..31
            const int ng = (tid & 7) << 3;      // 0..56
            const _Float16* bgp = Bm + (long)(k0 + kk) * ldb + n0 + ng;
            __align__(16) _Float16 tmp[8];
            if ((k0 + BK <= K) && (n0 + BN <= N)) {
                *(v8h*)tmp = *(const v8h*)bgp;
                __builtin_prefetch(bgp + (long)BK * ldb, 0, 1);
            } else {
                for (int j = 0; j < 8; ++j)
                    tmp[j] = ((k0 + kk) < K && (n0 + ng + j) < N) ? bgp[j] : (_Float16)0.f;
            }
            for (int j = 0; j < 8; ++j) Bs[(ng + j) * 40 + kk] = tmp[j];
        } else {      // B NxK row-major: rows are k-contiguous, direct copy
            const int nr = tid >> 2;            // 0..63
            const int kg = (tid & 3) << 3;      // 0..24
            const _Float16* bgp = Bm + (long)(n0 + nr) * ldb + k0 + kg;
            _Float16* blp = &Bs[nr * 40 + kg];
            if ((n0 + BN <= N) && (k0 + BK <= K)) {
#if HAVE_ASYNC
                async_cp16(blp, bgp);
#else
                *(v8h*)blp = *(const v8h*)bgp;
#endif
                __builtin_prefetch(bgp + BK, 0, 1);
            } else {
                for (int j = 0; j < 8; ++j)
                    blp[j] = ((n0 + nr) < N && (k0 + kg + j) < K) ? bgp[j] : (_Float16)0.f;
            }
        }
#if HAVE_ASYNC
        async_wait0();
#endif
        __syncthreads();

        // ---- fragments (A: lanes 0-15 = M rows, K halves split by lane group)
        const int kb  = (lane >> 4) << 3;
        const int mr  = wm * 32 + (lane & 15);
        v8h l0 = *(const v8h*)&As[mr * 40 + kb];
        v8h h0 = *(const v8h*)&As[mr * 40 + 16 + kb];
        v16h a0;
        #pragma unroll
        for (int j = 0; j < 8; ++j) { a0[j] = l0[j]; a0[j + 8] = h0[j]; }
        v8h l1 = *(const v8h*)&As[(mr + 16) * 40 + kb];
        v8h h1 = *(const v8h*)&As[(mr + 16) * 40 + 16 + kb];
        v16h a1;
        #pragma unroll
        for (int j = 0; j < 8; ++j) { a1[j] = l1[j]; a1[j + 8] = h1[j]; }

        const int kb2 = (lane >> 4) << 3;
        const int nr0 = wn * 32 + (lane & 15);
        v8h b0l = *(const v8h*)&Bs[nr0 * 40 + kb2];
        v8h b0h = *(const v8h*)&Bs[nr0 * 40 + kb2 + 16];
        v16h bf0;
        #pragma unroll
        for (int j = 0; j < 8; ++j) { bf0[j] = b0l[j]; bf0[j + 8] = b0h[j]; }
        v8h b1l = *(const v8h*)&Bs[(nr0 + 16) * 40 + kb2];
        v8h b1h = *(const v8h*)&Bs[(nr0 + 16) * 40 + kb2 + 16];
        v16h bf1;
        #pragma unroll
        for (int j = 0; j < 8; ++j) { bf1[j] = b1l[j]; bf1[j + 8] = b1h[j]; }

        c00 = __builtin_amdgcn_wmma_f32_16x16x32_f16(false, a0, false, bf0, (short)0, c00, false, false);
        c01 = __builtin_amdgcn_wmma_f32_16x16x32_f16(false, a0, false, bf1, (short)0, c01, false, false);
        c10 = __builtin_amdgcn_wmma_f32_16x16x32_f16(false, a1, false, bf0, (short)0, c10, false, false);
        c11 = __builtin_amdgcn_wmma_f32_16x16x32_f16(false, a1, false, bf1, (short)0, c11, false, false);
        __syncthreads();
    }

    // ---- epilogue per ISA C/D layout: VGPR j -> row (j | 8+j), lane%16 -> col
    const int mb = m0 + wm * 32 + ((lane >> 4) << 3);
    const int nb = n0 + wn * 32 + (lane & 15);
    #pragma unroll
    for (int j = 0; j < 8; ++j) {
        int mg = mb + j;
        if (mg < M) {
            if (nb      < N) C[(long)mg * ldc + nb     ] = c00[j];
            if (nb + 16 < N) C[(long)mg * ldc + nb + 16] = c01[j];
        }
        int mg2 = mb + 16 + j;
        if (mg2 < M) {
            if (nb      < N) C[(long)mg2 * ldc + nb     ] = c10[j];
            if (nb + 16 < N) C[(long)mg2 * ldc + nb + 16] = c11[j];
        }
    }
}

// ------------------------------------------- depthwise causal conv + SiLU
__global__ void k_conv_silu(const float* __restrict__ xz, const float* __restrict__ cw,
                            const float* __restrict__ cb, float* __restrict__ xc,
                            _Float16* __restrict__ xh) {
    long i = (long)blockIdx.x * blockDim.x + threadIdx.x;
    if (i >= (long)M_ * DI_) return;
    int d = (int)(i % DI_);
    long m = i / DI_;
    int l = (int)(m % L_);
    long rowbase = m * (2 * DI_);
    float acc = cb[d];
    #pragma unroll
    for (int j = 0; j < DC_; ++j) {
        int lp = l - (DC_ - 1) + j;
        if (lp >= 0)
            acc += cw[d * DC_ + j] * xz[rowbase + (long)(lp - l) * (2 * DI_) + d];
    }
    float s = acc / (1.f + __expf(-acc));
    xc[i] = s;
    xh[i] = (_Float16)s;
}

// ---------------------------------------------------- softplus(dt + b_dt)
__global__ void k_softplus(float* __restrict__ dt, const float* __restrict__ bdt, long n) {
    long i = (long)blockIdx.x * blockDim.x + threadIdx.x;
    if (i >= n) return;
    int d = (int)(i % DI_);
    float v = dt[i] + bdt[d];
    dt[i] = (v > 20.f) ? v : log1pf(__expf(v));
}

// ---- chunked associative scan: h_t = a_t*h_{t-1} + dt*x*B_t, y_t = C_t.h_t
// pass1: per (b,d,chunk) local scan (zero init): record decay product P and
//        local end-state H for each of the DS=16 diagonal states.
__global__ __launch_bounds__(256) void k_scan1(
        const float* __restrict__ dt, const float* __restrict__ xc,
        const float* __restrict__ xdb, const float* __restrict__ Alog,
        float* __restrict__ Pc, float* __restrict__ Hc) {
    __shared__ float bc[16];
    int dgrp = blockIdx.x % (DI_ / 256);
    int bcix = blockIdx.x / (DI_ / 256);   // b*NC_ + c
    int b = bcix / NC_, c = bcix % NC_;
    int d = dgrp * 256 + threadIdx.x;
    float Av[DS_];
    #pragma unroll
    for (int j = 0; j < DS_; ++j) Av[j] = -__expf(Alog[d * DS_ + j]);
    float P[DS_], H[DS_];
    #pragma unroll
    for (int j = 0; j < DS_; ++j) { P[j] = 1.f; H[j] = 0.f; }
    for (int t = c * LC_; t < (c + 1) * LC_; ++t) {
        long m = (long)b * L_ + t;
        __syncthreads();
        if (threadIdx.x < 16) bc[threadIdx.x] = xdb[m * XDBW_ + DR_ + threadIdx.x];
        __syncthreads();
        float dtt = dt[m * DI_ + d];
        float dx  = dtt * xc[m * DI_ + d];
        #pragma unroll
        for (int j = 0; j < DS_; ++j) {
            float a = __expf(dtt * Av[j]);
            P[j] *= a;
            H[j] = a * H[j] + dx * bc[j];
        }
    }
    long o = (((long)b * DI_ + d) * NC_ + c) * DS_;
    #pragma unroll
    for (int j = 0; j < DS_; ++j) { Pc[o + j] = P[j]; Hc[o + j] = H[j]; }
}

// pass2: per (b,d): stitch chunk states sequentially; emit start state per chunk.
__global__ void k_scan2(const float* __restrict__ Pc, const float* __restrict__ Hc,
                        float* __restrict__ Hs) {
    int i = blockIdx.x * blockDim.x + threadIdx.x;
    if (i >= B_ * DI_) return;
    float hs[DS_];
    #pragma unroll
    for (int j = 0; j < DS_; ++j) hs[j] = 0.f;
    long base = (long)i * NC_ * DS_;
    for (int c = 0; c < NC_; ++c) {
        long o = base + (long)c * DS_;
        #pragma unroll
        for (int j = 0; j < DS_; ++j) {
            Hs[o + j] = hs[j];
            hs[j] = Pc[o + j] * hs[j] + Hc[o + j];
        }
    }
}

// pass3: per (b,d,chunk) replay with correct init; fused (y+D*x)*silu(z) -> f16
__global__ __launch_bounds__(256) void k_scan3(
        const float* __restrict__ dt, const float* __restrict__ xc,
        const float* __restrict__ xz, const float* __restrict__ xdb,
        const float* __restrict__ Alog, const float* __restrict__ Dp,
        const float* __restrict__ Hs, _Float16* __restrict__ out) {
    __shared__ float bc[32];
    int dgrp = blockIdx.x % (DI_ / 256);
    int bcix = blockIdx.x / (DI_ / 256);
    int b = bcix / NC_, c = bcix % NC_;
    int d = dgrp * 256 + threadIdx.x;
    float Av[DS_];
    #pragma unroll
    for (int j = 0; j < DS_; ++j) Av[j] = -__expf(Alog[d * DS_ + j]);
    float Dpd = Dp[d];
    float st[DS_];
    long o = (((long)b * DI_ + d) * NC_ + c) * DS_;
    #pragma unroll
    for (int j = 0; j < DS_; ++j) st[j] = Hs[o + j];
    for (int t = c * LC_; t < (c + 1) * LC_; ++t) {
        long m = (long)b * L_ + t;
        __syncthreads();
        if (threadIdx.x < 32) bc[threadIdx.x] = xdb[m * XDBW_ + DR_ + threadIdx.x];
        __syncthreads();
        float dtt = dt[m * DI_ + d];
        float xt  = xc[m * DI_ + d];
        float zt  = xz[m * (2 * DI_) + DI_ + d];
        float dx = dtt * xt;
        float y = 0.f;
        #pragma unroll
        for (int j = 0; j < DS_; ++j) {
            st[j] = __expf(dtt * Av[j]) * st[j] + dx * bc[j];
            y += st[j] * bc[DS_ + j];
        }
        float yo = y + Dpd * xt;
        float sz = zt / (1.f + __expf(-zt));
        out[m * DI_ + d] = (_Float16)(yo * sz);
    }
}

// ------------------------------------------------------------ residual add
__global__ void k_add(float* __restrict__ x, const float* __restrict__ y, long n) {
    long i = (long)blockIdx.x * blockDim.x + threadIdx.x;
    if (i < n) x[i] += y[i];
}

extern "C" void kernel_launch(void* const* d_in, const int* in_sizes, int n_in,
                              void* d_out, int out_size, void* d_ws, size_t ws_size,
                              hipStream_t stream) {
    (void)in_sizes; (void)n_in; (void)out_size; (void)ws_size;
    const int*   ids    = (const int*)  d_in[0];
    const float* embed  = (const float*)d_in[1];
    const float* ln_g   = (const float*)d_in[2];
    const float* ln_b   = (const float*)d_in[3];
    const float* W_in   = (const float*)d_in[4];
    const float* conv_w = (const float*)d_in[5];
    const float* conv_b = (const float*)d_in[6];
    const float* W_xp   = (const float*)d_in[7];
    const float* W_dt   = (const float*)d_in[8];
    const float* b_dt   = (const float*)d_in[9];
    const float* A_log  = (const float*)d_in[10];
    const float* D_p    = (const float*)d_in[11];
    const float* W_out  = (const float*)d_in[12];
    const float* fn_g   = (const float*)d_in[13];
    const float* fn_b   = (const float*)d_in[14];

    char* p = (char*)d_ws;
    auto alloc = [&](size_t bytes) -> void* {
        void* r = (void*)p;
        p += (bytes + 255) & ~(size_t)255;
        return r;
    };
    float*    x_res = (float*)   alloc((size_t)M_ * D_ * 4);
    _Float16* h16   = (_Float16*)alloc((size_t)M_ * DI_ * 2);
    _Float16* wbuf  = (_Float16*)alloc((size_t)D_ * 2 * DI_ * 2);
    float*    xz    = (float*)   alloc((size_t)M_ * 2 * DI_ * 4);
    float*    xc    = (float*)   alloc((size_t)M_ * DI_ * 4);
    float*    xdb   = (float*)   alloc((size_t)M_ * XDBW_ * 4);
    _Float16* dt16  = (_Float16*)alloc((size_t)M_ * DR_ * 2);
    float*    dtb   = (float*)   alloc((size_t)M_ * DI_ * 4);
    float*    yproj = (float*)   alloc((size_t)M_ * D_ * 4);
    float*    Pc    = (float*)   alloc((size_t)B_ * DI_ * NC_ * DS_ * 4);
    float*    Hc    = (float*)   alloc((size_t)B_ * DI_ * NC_ * DS_ * 4);
    float*    Hs    = (float*)   alloc((size_t)B_ * DI_ * NC_ * DS_ * 4);
    _Float16* emb16 = (_Float16*)alloc((size_t)V_ * D_ * 2);

    dim3 blk(256);
    auto cdiv = [](long a, long b) { return (int)((a + b - 1) / b); };

    k_gather<<<cdiv((long)M_ * D_, 256), blk, 0, stream>>>(ids, embed, x_res, M_ * D_);

    for (int i = 0; i < NL_; ++i) {
        const float* Wi  = W_in   + (size_t)i * D_  * 2 * DI_;
        const float* cwi = conv_w + (size_t)i * DI_ * DC_;
        const float* cbi = conv_b + (size_t)i * DI_;
        const float* Wxp = W_xp   + (size_t)i * DI_ * XDBW_;
        const float* Wdt = W_dt   + (size_t)i * DR_ * DI_;
        const float* bdt = b_dt   + (size_t)i * DI_;
        const float* Al  = A_log  + (size_t)i * DI_ * DS_;
        const float* Dpi = D_p    + (size_t)i * DI_;
        const float* Wo  = W_out  + (size_t)i * DI_ * D_;

        k_ln16<<<M_, 256, 0, stream>>>(x_res, ln_g + (size_t)i * D_, ln_b + (size_t)i * D_, h16);
        k_cast16<<<cdiv((long)D_ * 2 * DI_, 256), blk, 0, stream>>>(Wi, wbuf, D_, 2 * DI_, 2 * DI_, 2 * DI_);
        k_gemm_wmma<<<dim3(cdiv(2 * DI_, BN), cdiv(M_, BM)), blk, 0, stream>>>(
            h16, wbuf, xz, M_, 2 * DI_, D_, D_, 2 * DI_, 2 * DI_, 0);

        k_conv_silu<<<cdiv((long)M_ * DI_, 256), blk, 0, stream>>>(xz, cwi, cbi, xc, h16);

        k_cast16<<<cdiv((long)DI_ * XDBW_, 256), blk, 0, stream>>>(Wxp, wbuf, DI_, XDBW_, XDBW_, XDBW_);
        k_gemm_wmma<<<dim3(cdiv(XDBW_, BN), cdiv(M_, BM)), blk, 0, stream>>>(
            h16, wbuf, xdb, M_, XDBW_, DI_, DI_, XDBW_, XDBW_, 0);

        k_cast16<<<cdiv((long)M_ * DR_, 256), blk, 0, stream>>>(xdb, dt16, M_, DR_, XDBW_, DR_);
        k_cast16<<<cdiv((long)DR_ * DI_, 256), blk, 0, stream>>>(Wdt, wbuf, DR_, DI_, DI_, DI_);
        k_gemm_wmma<<<dim3(cdiv(DI_, BN), cdiv(M_, BM)), blk, 0, stream>>>(
            dt16, wbuf, dtb, M_, DI_, DR_, DR_, DI_, DI_, 0);
        k_softplus<<<cdiv((long)M_ * DI_, 256), blk, 0, stream>>>(dtb, bdt, (long)M_ * DI_);

        k_scan1<<<B_ * NC_ * (DI_ / 256), blk, 0, stream>>>(dtb, xc, xdb, Al, Pc, Hc);
        k_scan2<<<cdiv(B_ * DI_, 256), blk, 0, stream>>>(Pc, Hc, Hs);
        k_scan3<<<B_ * NC_ * (DI_ / 256), blk, 0, stream>>>(dtb, xc, xz, xdb, Al, Dpi, Hs, h16);

        k_cast16<<<cdiv((long)DI_ * D_, 256), blk, 0, stream>>>(Wo, wbuf, DI_, D_, D_, D_);
        k_gemm_wmma<<<dim3(cdiv(D_, BN), cdiv(M_, BM)), blk, 0, stream>>>(
            h16, wbuf, yproj, M_, D_, DI_, DI_, D_, D_, 0);
        k_add<<<cdiv((long)M_ * D_, 256), blk, 0, stream>>>(x_res, yproj, (long)M_ * D_);
    }

    k_ln16<<<M_, 256, 0, stream>>>(x_res, fn_g, fn_b, h16);
    k_cast16<<<cdiv((long)V_ * D_, 256), blk, 0, stream>>>(embed, emb16, V_, D_, D_, D_);
    // logits: A = final hidden (2048x768 f16), B = embed rows as B^T (b_nt=1)
    k_gemm_wmma<<<dim3(cdiv(V_, BN), cdiv(M_, BM)), blk, 0, stream>>>(
        h16, emb16, (float*)d_out, M_, V_, D_, D_, D_, V_, 1);
}
